// CBAM_83691732730338
// MI455X (gfx1250) — compile-verified, matrix operand order checked by reading, not run on hardware
//
#include <hip/hip_runtime.h>
#include <hip/hip_bf16.h>

// ---------------------------------------------------------------------------
// Problem constants (B=4, C=64, H=W=64, D=4 -> N = 4096, CQ*D = 32, C*D = 256)
// ---------------------------------------------------------------------------
#define NB   4
#define NC   64
#define NN   4096      // H*W
#define ND   4
#define KQ   32        // CQ*D  (q/k channel dim)
#define CD   256       // C*D   (v channel dim)

typedef __attribute__((ext_vector_type(16))) __bf16 v16bf;
typedef __attribute__((ext_vector_type(8)))  __bf16 v8bf;
typedef __attribute__((ext_vector_type(8)))  float  v8f;

union FragBF {
    v16bf v;
    v8bf  h[2];
};

// ---------------------------------------------------------------------------
// DPP16-based 16-lane reductions (no LDS, no waits; stays within lane halves)
//   quad_perm xor1 = 0xB1, quad_perm xor2 = 0x4E,
//   row_half_mirror = 0x141, row_mirror = 0x140
// ---------------------------------------------------------------------------
template <int CTRL>
__device__ __forceinline__ float dpp_bcast(float x) {
    int xi = __builtin_bit_cast(int, x);
    int yi = __builtin_amdgcn_update_dpp(0, xi, CTRL, 0xf, 0xf, true);
    return __builtin_bit_cast(float, yi);
}
__device__ __forceinline__ float red16_max(float x) {
    x = fmaxf(x, dpp_bcast<0x0B1>(x));
    x = fmaxf(x, dpp_bcast<0x04E>(x));
    x = fmaxf(x, dpp_bcast<0x141>(x));
    x = fmaxf(x, dpp_bcast<0x140>(x));
    return x;
}
__device__ __forceinline__ float red16_sum(float x) {
    x += dpp_bcast<0x0B1>(x);
    x += dpp_bcast<0x04E>(x);
    x += dpp_bcast<0x141>(x);
    x += dpp_bcast<0x140>(x);
    return x;
}

// ---------------------------------------------------------------------------
// Kernel 1: QKV projection (1x1 conv == channel matmul), writes bf16:
//   Qb[b][n][kq]   (kq contiguous -> A-fragment friendly)
//   Kb[b][m][kq]   (kq contiguous -> B-fragment friendly for energy)
//   Vt[b][cd][m]   (m contiguous  -> B-fragment friendly for P*V)
// ---------------------------------------------------------------------------
__global__ __launch_bounds__(256) void qkv_kernel(
    const float* __restrict__ x2d,
    const float* __restrict__ Wq, const float* __restrict__ bq,
    const float* __restrict__ Wk, const float* __restrict__ bk,
    const float* __restrict__ Wv, const float* __restrict__ bv,
    __bf16* __restrict__ Qb, __bf16* __restrict__ Kb, __bf16* __restrict__ Vt)
{
    const int b  = blockIdx.x >> 7;           // 128 n-chunks per batch
    const int n0 = (blockIdx.x & 127) * 32;
    const int t  = threadIdx.x;

    __shared__ float xs[NC * ND * 32];        // [c][d][nl] : 32 KB

    const float* xb = x2d + (size_t)b * NC * NN * ND;
    #pragma unroll 4
    for (int i = t; i < NC * ND * 32; i += 256) {
        int c   = i >> 7;
        int rem = i & 127;
        int nl  = rem >> 2;
        int d   = rem & 3;
        xs[(c * 4 + d) * 32 + nl] = xb[((size_t)c * NN + n0 + nl) * ND + d];
    }
    __syncthreads();

    // --- V projection: thread t owns output channel cd = t (256 channels)
    {
        const int cd = t, c = cd >> 2, d = cd & 3;
        float acc[32];
        const float bias = bv[c];
        #pragma unroll
        for (int n = 0; n < 32; ++n) acc[n] = bias;
        for (int c2 = 0; c2 < NC; ++c2) {
            const float w = Wv[c * NC + c2];
            const float* xr = &xs[(c2 * 4 + d) * 32];
            #pragma unroll
            for (int n = 0; n < 32; ++n) acc[n] += w * xr[n];
        }
        __bf16* dst = Vt + (size_t)(b * CD + cd) * NN + n0;
        #pragma unroll
        for (int n = 0; n < 32; ++n) dst[n] = (__bf16)acc[n];
    }

    // --- Q (wave 0) and K (wave 1) projections: 32 channels each
    if (t < 64) {
        const int isK = t >> 5;
        const int kq  = t & 31, cq = kq >> 2, d = kq & 3;
        const float* Wm   = isK ? Wk : Wq;
        const float  bias = isK ? bk[cq] : bq[cq];
        float acc[32];
        #pragma unroll
        for (int n = 0; n < 32; ++n) acc[n] = bias;
        for (int c2 = 0; c2 < NC; ++c2) {
            const float w = Wm[cq * NC + c2];
            const float* xr = &xs[(c2 * 4 + d) * 32];
            #pragma unroll
            for (int n = 0; n < 32; ++n) acc[n] += w * xr[n];
        }
        __bf16* dst = (isK ? Kb : Qb) + ((size_t)b * NN + n0) * KQ + kq;
        #pragma unroll
        for (int n = 0; n < 32; ++n) dst[(size_t)n * KQ] = (__bf16)acc[n];
    }
}

// ---------------------------------------------------------------------------
// Kernel 2: streaming-softmax attention. One wave32 owns a 16-row tile of N
// and the full 256-wide output, iterating m in blocks of 64.
// ---------------------------------------------------------------------------
#define PROW 72   // P row stride in bf16 elems (64 + 8 pad -> conflict-free TR)

__global__ __launch_bounds__(128) void attn_kernel(
    const __bf16* __restrict__ Qb, const __bf16* __restrict__ Kb,
    const __bf16* __restrict__ Vt, const float* __restrict__ x3d,
    const float* __restrict__ gamma, float* __restrict__ out)
{
    __shared__ __bf16 Pbuf[4 * 16 * PROW];

    const int lane = threadIdx.x & 31;
    const int wv   = threadIdx.x >> 5;
    const int wg   = blockIdx.x * 4 + wv;     // 1024 waves total
    const int b     = wg >> 8;                // 256 row-tiles per batch
    const int ntile = (wg & 255) << 4;
    const int l15   = lane & 15;
    const int h     = lane >> 4;

    __bf16* P = &Pbuf[wv * 16 * PROW];

    // Q fragment for this 16-row tile: A layout, K chunks at 8h and 16+8h
    FragBF qf;
    {
        const __bf16* qp = Qb + ((size_t)b * NN + ntile + l15) * KQ + 8 * h;
        qf.h[0] = *(const v8bf*)(qp);
        qf.h[1] = *(const v8bf*)(qp + 16);
    }

    v8f acc[16];
    #pragma unroll
    for (int cf = 0; cf < 16; ++cf)
        #pragma unroll
        for (int j = 0; j < 8; ++j) acc[cf][j] = 0.0f;

    float m_i[8], l_i[8];
    #pragma unroll
    for (int j = 0; j < 8; ++j) { m_i[j] = -__builtin_inff(); l_i[j] = 0.0f; }

    const __bf16* Kbase = Kb + (size_t)b * NN * KQ;
    const __bf16* Vbase = Vt + (size_t)b * CD * NN;

    for (int mb = 0; mb < NN / 64; ++mb) {
        const int m0 = mb * 64;

        // ---- prefetch next m-block's K (4KB) and V (32KB) cachelines
        if (mb + 1 < NN / 64) {
            const int m1 = m0 + 64;
            __builtin_prefetch(Kbase + (size_t)m1 * KQ + lane * 64, 0, 0);
            #pragma unroll
            for (int pv = 0; pv < 8; ++pv)
                __builtin_prefetch(Vbase + (size_t)(pv * 32 + lane) * NN + m1,
                                   0, 0);
        }

        // ---- load ALL K fragments first (one burst), then 4 energy WMMAs
        FragBF kf[4];
        #pragma unroll
        for (int mf = 0; mf < 4; ++mf) {
            const __bf16* kp = Kbase + (size_t)(m0 + mf * 16 + l15) * KQ + 8 * h;
            kf[mf].h[0] = *(const v8bf*)(kp);
            kf[mf].h[1] = *(const v8bf*)(kp + 16);
        }
        v8f e[4];
        #pragma unroll
        for (int mf = 0; mf < 4; ++mf) {
            v8f z = {};
            e[mf] = __builtin_amdgcn_wmma_f32_16x16x32_bf16(
                        false, qf.v, false, kf[mf].v, (short)0, z, false, false);
        }

        // ---- per-row block max via DPP16 (no LDS)
        float bm[8];
        #pragma unroll
        for (int j = 0; j < 8; ++j) {
            float v = fmaxf(fmaxf(e[0][j], e[1][j]), fmaxf(e[2][j], e[3][j]));
            bm[j] = fmaxf(red16_max(v), m_i[j]);
        }

        // ---- online rescale of accumulator + denominator
        #pragma unroll
        for (int j = 0; j < 8; ++j) {
            const float s = __expf(m_i[j] - bm[j]);
            m_i[j] = bm[j];
            l_i[j] *= s;
            #pragma unroll
            for (int cf = 0; cf < 16; ++cf) acc[cf][j] *= s;
        }

        // ---- P = exp(e - m): fp32 stats in regs, bf16 tile to LDS (D->A)
        #pragma unroll
        for (int mf = 0; mf < 4; ++mf) {
            #pragma unroll
            for (int j = 0; j < 8; ++j) {
                const float p = __expf(e[mf][j] - m_i[j]);
                l_i[j] += p;                       // per-lane partial row sum
                P[(8 * h + j) * PROW + mf * 16 + l15] = (__bf16)p;
            }
        }
        // Same-wave DS ops are in-order: no barrier needed before reload.

        // ---- acc += P(16x64) * V^T(64x256), V double-buffered
        #pragma unroll
        for (int ms = 0; ms < 2; ++ms) {
            FragBF pf;
            const __bf16* pp = P + (size_t)l15 * PROW + ms * 32 + 8 * h;
            pf.h[0] = *(const v8bf*)(pp);
            pf.h[1] = *(const v8bf*)(pp + 16);

            const __bf16* vrow = Vbase + (size_t)l15 * NN + m0 + ms * 32 + 8 * h;
            FragBF vf[2];
            vf[0].h[0] = *(const v8bf*)(vrow);
            vf[0].h[1] = *(const v8bf*)(vrow + 16);
            #pragma unroll
            for (int cf = 0; cf < 16; ++cf) {
                if (cf < 15) {
                    const __bf16* vp = vrow + (size_t)(cf + 1) * 16 * NN;
                    vf[(cf + 1) & 1].h[0] = *(const v8bf*)(vp);
                    vf[(cf + 1) & 1].h[1] = *(const v8bf*)(vp + 16);
                }
                acc[cf] = __builtin_amdgcn_wmma_f32_16x16x32_bf16(
                            false, pf.v, false, vf[cf & 1].v, (short)0,
                            acc[cf], false, false);
            }
        }
    }

    // ---- finalize: reduce denominators across 16-lane halves (DPP16)
    #pragma unroll
    for (int j = 0; j < 8; ++j)
        l_i[j] = 1.0f / red16_sum(l_i[j]);

    // ---- epilogue: out = gamma * (acc / l) + x3d, back in (B,C,H,W,D)
    const float g = gamma[0];
    #pragma unroll
    for (int cf = 0; cf < 16; ++cf) {
        const int cd = cf * 16 + l15;
        const int cc = cd >> 2, d = cd & 3;
        #pragma unroll
        for (int j = 0; j < 8; ++j) {
            const int n = ntile + 8 * h + j;
            const size_t idx = (((size_t)(b * NC + cc) * NN) + n) * ND + d;
            out[idx] = g * acc[cf][j] * l_i[j] + x3d[idx];
        }
    }
}

// ---------------------------------------------------------------------------
// Host-side launcher
// ---------------------------------------------------------------------------
extern "C" void kernel_launch(void* const* d_in, const int* in_sizes, int n_in,
                              void* d_out, int out_size, void* d_ws, size_t ws_size,
                              hipStream_t stream)
{
    const float* x2d   = (const float*)d_in[0];
    const float* x3d   = (const float*)d_in[1];
    const float* Wq    = (const float*)d_in[2];
    const float* bq    = (const float*)d_in[3];
    const float* Wk    = (const float*)d_in[4];
    const float* bk    = (const float*)d_in[5];
    const float* Wv    = (const float*)d_in[6];
    const float* bv    = (const float*)d_in[7];
    const float* gamma = (const float*)d_in[8];
    float* out = (float*)d_out;

    // Workspace: Qb (1MB) | Kb (1MB) | Vt (8MB), all bf16
    __bf16* Qb = (__bf16*)d_ws;
    __bf16* Kb = Qb + (size_t)NB * NN * KQ;
    __bf16* Vt = Kb + (size_t)NB * NN * KQ;

    qkv_kernel<<<dim3(NB * (NN / 32)), dim3(256), 0, stream>>>(
        x2d, Wq, bq, Wk, bk, Wv, bv, Qb, Kb, Vt);

    attn_kernel<<<dim3((NB * (NN / 16)) / 4), dim3(128), 0, stream>>>(
        Qb, Kb, Vt, x3d, gamma, out);
}
